// VariationalGCNEncoder_9801115369952
// MI455X (gfx1250) — matrix operand order repaired and checked by reading, not compiled, not run on hardware
//
#include <hip/hip_runtime.h>
#include <hip/hip_bf16.h>

typedef __attribute__((ext_vector_type(16))) _Float16 v16h;
typedef __attribute__((ext_vector_type(8)))  float    v8f;

#define IN_F  256
#define HID_F 64
#define OUT_F 32

// ---------------------------------------------------------------- utilities
__global__ void fill_kernel(float* __restrict__ p, float v, long long n) {
    long long i = (long long)blockIdx.x * blockDim.x + threadIdx.x;
    if (i < n) p[i] = v;
}

// deg pre-initialized to 1.0 (self-loop contribution)
__global__ void degree_kernel(const long long* __restrict__ dst,
                              float* __restrict__ deg, int E) {
    int e = blockIdx.x * blockDim.x + threadIdx.x;
    if (e < E) atomicAdd(&deg[(int)dst[e]], 1.0f);
}

__global__ void dinv_kernel(float* __restrict__ deg, int n) {
    int i = blockIdx.x * blockDim.x + threadIdx.x;
    if (i < n) {
        float d = deg[i];
        deg[i] = (d > 0.0f) ? rsqrtf(d) : 0.0f;
    }
}

// ---------------------------------------------------------------- B pre-swizzle
// Convert f32 weights into the exact CDNA5 WMMA 16-bit B-operand layout so the
// GEMM can load each lane's 16-element fragment as one contiguous 32B v16h.
// Layout: Bsw[((ks*4 + coltile)*32 + lane)*16 + e]
//   lane: half = lane>>4 selects K group ({0..7,16..23} vs {8..15,24..31} of the 32-K step),
//         m = lane&15 selects column within the 16-wide tile.
__global__ void swizzleB1_kernel(const float* __restrict__ W, _Float16* __restrict__ Bsw) {
    int t = blockIdx.x * blockDim.x + threadIdx.x;   // 8 ksteps * 4 coltiles * 32 lanes
    if (t >= 8 * 4 * 32) return;
    int lane = t & 31;
    int ct   = (t >> 5) & 3;
    int ks   = t >> 7;
    int half = lane >> 4, m = lane & 15;
    int col  = ct * 16 + m;
    #pragma unroll
    for (int e = 0; e < 16; ++e)
        Bsw[(long long)t * 16 + e] = (_Float16)W[(long long)(ks * 32 + 8 * half + (e < 8 ? e : e + 8)) * HID_F + col];
}

// Fused [Wmu | Wlv] (64x32 each) -> 64x64 B in swizzled layout; 2 ksteps.
__global__ void swizzleB2_kernel(const float* __restrict__ Wmu, const float* __restrict__ Wlv,
                                 _Float16* __restrict__ Bsw) {
    int t = blockIdx.x * blockDim.x + threadIdx.x;   // 2 * 4 * 32
    if (t >= 2 * 4 * 32) return;
    int lane = t & 31;
    int ct   = (t >> 5) & 3;
    int ks   = t >> 7;
    int half = lane >> 4, m = lane & 15;
    int col  = ct * 16 + m;
    const float* W = (col < OUT_F) ? Wmu : Wlv;
    int c = (col < OUT_F) ? col : col - OUT_F;
    #pragma unroll
    for (int e = 0; e < 16; ++e)
        Bsw[(long long)t * 16 + e] = (_Float16)W[(long long)(ks * 32 + 8 * half + (e < 8 ? e : e + 8)) * OUT_F + c];
}

// ---------------------------------------------------------------- WMMA GEMM
// C[nrows,64] = A[nrows,K] @ B[K,64] with pre-swizzled f16 B, f32 accumulate.
// 128 threads = 4 waves; wave w owns cols [16w,16w+16) of a 16-row stripe.
template <int K>
__global__ void __launch_bounds__(128)
gemm_wmma_kernel(const float* __restrict__ A, const _Float16* __restrict__ Bsw,
                 float* __restrict__ C, int nrows) {
    const int lane = threadIdx.x & 31;
    const int wave = threadIdx.x >> 5;
    const int m    = lane & 15;
    const int half = lane >> 4;
    const int rowA = min((int)blockIdx.x * 16 + m, nrows - 1);  // clamp: EXEC stays full
    const int col  = wave * 16 + m;

    v8f acc = {};
    #pragma unroll
    for (int ks = 0; ks < K / 32; ++ks) {
        // A fragment: lane half 0 -> K {0..7,16..23}; half 1 -> K {8..15,24..31}
        const float4* A4 = (const float4*)(A + (long long)rowA * K + ks * 32);
        float4 p0 = A4[2 * half];
        float4 p1 = A4[2 * half + 1];
        float4 p2 = A4[4 + 2 * half];
        float4 p3 = A4[5 + 2 * half];
        v16h a;
        a[0]  = (_Float16)p0.x; a[1]  = (_Float16)p0.y; a[2]  = (_Float16)p0.z; a[3]  = (_Float16)p0.w;
        a[4]  = (_Float16)p1.x; a[5]  = (_Float16)p1.y; a[6]  = (_Float16)p1.z; a[7]  = (_Float16)p1.w;
        a[8]  = (_Float16)p2.x; a[9]  = (_Float16)p2.y; a[10] = (_Float16)p2.z; a[11] = (_Float16)p2.w;
        a[12] = (_Float16)p3.x; a[13] = (_Float16)p3.y; a[14] = (_Float16)p3.z; a[15] = (_Float16)p3.w;
        // B fragment: one contiguous 32B load in WMMA layout
        v16h b = *(const v16h*)(Bsw + (((long long)ks * 4 + wave) * 32 + lane) * 16);
        acc = __builtin_amdgcn_wmma_f32_16x16x32_f16(false, a, false, b,
                                                     (short)0, acc, false, false);
    }
    // D: VGPR r holds M=r (lanes 0-15) and M=r+8 (lanes 16-31)
    const int base = (int)blockIdx.x * 16;
    if (base + 16 <= nrows) {                       // uniform fast path: branchless stores
        float* crow = C + (long long)base * HID_F + (long long)8 * half * HID_F + col;
        #pragma unroll
        for (int r = 0; r < 8; ++r)
            crow[(long long)r * HID_F] = acc[r];
    } else {
        #pragma unroll
        for (int r = 0; r < 8; ++r) {
            int row = base + r + 8 * half;
            if (row < nrows) C[(long long)row * HID_F + col] = acc[r];
        }
    }
}

// ------------------------- conv1 aggregation: one wave per edge, scalar indices
__global__ void __launch_bounds__(256)
agg1_kernel(const long long* __restrict__ src, const long long* __restrict__ dst,
            const float* __restrict__ dinv, const float* __restrict__ xw,
            float* __restrict__ hbuf, int E) {
    int ew = __builtin_amdgcn_readfirstlane(
        (int)blockIdx.x * ((int)blockDim.x >> 5) + ((int)threadIdx.x >> 5));
    if (ew >= E) return;                              // uniform
    int lane = threadIdx.x & 31;
    int s = (int)src[ew], d = (int)dst[ew];
    float w = dinv[s] * dinv[d];
    const float* xs = xw + (long long)s * HID_F;
    float* hd = hbuf + (long long)d * HID_F;
    atomicAdd(&hd[lane],      xs[lane]      * w);
    atomicAdd(&hd[lane + 32], xs[lane + 32] * w);
}

// h = relu(hbuf + selfloop + b1); in place over hbuf
__global__ void relu_bias_kernel(const float* __restrict__ xw, const float* __restrict__ b1,
                                 const float* __restrict__ dinv, float* __restrict__ h,
                                 long long total) {
    long long t = (long long)blockIdx.x * blockDim.x + threadIdx.x;
    if (t >= total) return;
    int i = (int)(t >> 6);
    int f = (int)(t & 63);
    float di = dinv[i];
    float v = h[t] + xw[t] * di * di + b1[f];
    h[t] = fmaxf(v, 0.0f);
}

// ---------- conv2 aggregation: lane covers mu (f=lane) and lv (f=lane+32)
__global__ void __launch_bounds__(256)
agg2_kernel(const long long* __restrict__ src, const long long* __restrict__ dst,
            const float* __restrict__ dinv, const float* __restrict__ hw,
            float* __restrict__ out, int Nn, int E) {
    int ew = __builtin_amdgcn_readfirstlane(
        (int)blockIdx.x * ((int)blockDim.x >> 5) + ((int)threadIdx.x >> 5));
    if (ew >= E) return;
    int lane = threadIdx.x & 31;
    int s = (int)src[ew], d = (int)dst[ew];
    float w = dinv[s] * dinv[d];
    const float* hs = hw + (long long)s * HID_F;
    atomicAdd(&out[(long long)d * OUT_F + lane], hs[lane] * w);
    atomicAdd(&out[(long long)Nn * OUT_F + (long long)d * OUT_F + lane], hs[lane + 32] * w);
}

// add self-loop term + biases
__global__ void final_kernel(const float* __restrict__ hw, const float* __restrict__ dinv,
                             const float* __restrict__ bmu, const float* __restrict__ blv,
                             float* __restrict__ out, int Nn, long long total) {
    long long t = (long long)blockIdx.x * blockDim.x + threadIdx.x;
    if (t >= total) return;
    int i = (int)(t >> 6);
    int f = (int)(t & 63);
    float di = dinv[i];
    float v = hw[t] * di * di;
    if (f < OUT_F) out[(long long)i * OUT_F + f] += v + bmu[f];
    else out[(long long)Nn * OUT_F + (long long)i * OUT_F + (f - OUT_F)] += v + blv[f - OUT_F];
}

// ---------------------------------------------------------------- launch
extern "C" void kernel_launch(void* const* d_in, const int* in_sizes, int n_in,
                              void* d_out, int out_size, void* d_ws, size_t ws_size,
                              hipStream_t stream) {
    const float*     x   = (const float*)d_in[0];
    const long long* ei  = (const long long*)d_in[1];   // int64 in reference
    const float*     W1  = (const float*)d_in[2];
    const float*     b1  = (const float*)d_in[3];
    const float*     Wmu = (const float*)d_in[4];
    const float*     bmu = (const float*)d_in[5];
    const float*     Wlv = (const float*)d_in[6];
    const float*     blv = (const float*)d_in[7];
    float* out = (float*)d_out;

    const int N = in_sizes[0] / IN_F;
    const int E = in_sizes[1] / 2;
    const long long* src = ei;
    const long long* dst = ei + E;

    // workspace layout (256B-aligned sections)
    float* dinv = (float*)d_ws;                               // N
    float* xw   = dinv + (((long long)N + 63) & ~63LL);       // N*64 (reused as hw)
    float* hbuf = xw + (long long)N * HID_F;                  // N*64 (reused as h)
    _Float16* Bsw1 = (_Float16*)(hbuf + (long long)N * HID_F);// 8*4*32*16 f16 (32KB)
    _Float16* Bsw2 = Bsw1 + 8 * 4 * 32 * 16;                  // 2*4*32*16 f16 (8KB)

    auto cdiv = [](long long a, long long b) { return (unsigned)((a + b - 1) / b); };
    const long long totN = (long long)N * HID_F;

    // init + weight swizzle (independent, launched up front)
    fill_kernel<<<cdiv(N, 256), 256, 0, stream>>>(dinv, 1.0f, N);       // self-loop degree
    fill_kernel<<<cdiv(totN, 256), 256, 0, stream>>>(hbuf, 0.0f, totN);
    fill_kernel<<<cdiv(out_size, 256), 256, 0, stream>>>(out, 0.0f, out_size);
    swizzleB1_kernel<<<4, 256, 0, stream>>>(W1, Bsw1);
    swizzleB2_kernel<<<1, 256, 0, stream>>>(Wmu, Wlv, Bsw2);

    // normalization
    degree_kernel<<<cdiv(E, 256), 256, 0, stream>>>(dst, dinv, E);
    dinv_kernel<<<cdiv(N, 256), 256, 0, stream>>>(dinv, N);

    // conv1: dense (WMMA) then sparse scatter
    gemm_wmma_kernel<IN_F><<<cdiv(N, 16), 128, 0, stream>>>(x, Bsw1, xw, N);
    agg1_kernel<<<cdiv(E, 8), 256, 0, stream>>>(src, dst, dinv, xw, hbuf, E);
    relu_bias_kernel<<<cdiv(totN, 256), 256, 0, stream>>>(xw, b1, dinv, hbuf, totN);

    // conv2 (mu & logvar fused): dense (WMMA) then sparse; hw reuses xw region
    gemm_wmma_kernel<HID_F><<<cdiv(N, 16), 128, 0, stream>>>(hbuf, Bsw2, xw, N);
    agg2_kernel<<<cdiv(E, 8), 256, 0, stream>>>(src, dst, dinv, xw, out, N, E);
    final_kernel<<<cdiv(totN, 256), 256, 0, stream>>>(xw, dinv, bmu, blv, out, N, totN);
}
